// Model_60052232732758
// MI455X (gfx1250) — compile-verified, matrix operand order checked by reading, not run on hardware
//
#include <hip/hip_runtime.h>

#define N_NODES_C   100000
#define EMB_C       256
#define N_MP_E      1600000
#define N_SUP_E     200000

typedef __attribute__((ext_vector_type(16))) __bf16 v16bf;
typedef __attribute__((ext_vector_type(8)))  float  v8f;
typedef __attribute__((ext_vector_type(4)))  unsigned int v4u;
typedef __attribute__((ext_vector_type(8)))  int v8i;
typedef __attribute__((ext_vector_type(4)))  int v4i;

union AFrag { v16bf v; unsigned int u[8]; };
union CFrag { v8f   v; float        f[8]; };

__device__ __forceinline__ unsigned short f2bf(float f) {
  unsigned int u = __builtin_bit_cast(unsigned int, f);
  u += 0x7FFFu + ((u >> 16) & 1u);           // round-to-nearest-even
  return (unsigned short)(u >> 16);
}

// ---------------- weight / activation conversion ----------------
__global__ void cvt_f32_bf16(const float* __restrict__ in,
                             unsigned short* __restrict__ out, int n) {
  int i = blockIdx.x * blockDim.x + threadIdx.x;
  if (i < n) out[i] = f2bf(in[i]);
}

// ---------------- scatter-add (segment sum + counts) ----------------
__global__ void sage_scatter(const float* __restrict__ x,
                             const int* __restrict__ src,
                             const int* __restrict__ dst,
                             float* __restrict__ sum,
                             float* __restrict__ cnt) {
  const int EPB = 8;
  int e0 = blockIdx.x * EPB;
  int t  = threadIdx.x;                      // channel 0..255
  #pragma unroll
  for (int i = 0; i < EPB; ++i) {
    int e = e0 + i;
    if (e >= N_MP_E) return;
    int s = src[e];
    int d = dst[e];
    if (e + 1 < N_MP_E) {                    // prefetch next source row
      __builtin_prefetch(&x[(size_t)src[e + 1] * EMB_C + t], 0, 1);
    }
    float v = x[(size_t)s * EMB_C + t];
    atomicAdd(&sum[(size_t)d * EMB_C + t], v);
    if (t == 0) atomicAdd(&cnt[d], 1.0f);
  }
}

// ---------------- mean = sum / max(cnt,1); stage bf16 copies ----------------
__global__ void mean_cvt(const float* __restrict__ sum,
                         const float* __restrict__ cnt,
                         const float* __restrict__ xin,
                         unsigned short* __restrict__ meanB,
                         unsigned short* __restrict__ xB) {
  size_t i = (size_t)blockIdx.x * blockDim.x + threadIdx.x;
  if (i >= (size_t)N_NODES_C * EMB_C) return;
  int n = (int)(i >> 8);
  float c = fmaxf(cnt[n], 1.0f);
  meanB[i] = f2bf(sum[i] / c);
  xB[i]    = f2bf(xin[i]);
}

// ---------------- TDM: DMA a 16x256 bf16 tile (rows of W) into LDS ----------
__device__ __forceinline__ void tdm_load_16x256_bf16(const unsigned short* gsrc,
                                                     unsigned lds_byte_off) {
  unsigned long long ga = (unsigned long long)gsrc;
  // D# group0: count=1 | lds_addr | global_addr[56:0] | type=2
  v4u g0 = { 1u,
             lds_byte_off,
             (unsigned)(ga & 0xFFFFFFFFu),
             (unsigned)((ga >> 32) & 0x01FFFFFFu) | (2u << 30) };
  // D# group1: data_size=2B; tensor 256x256; tile 256x16; dim0 stride 256
  v8i g1 = { (int)0x00010000,   // workgroup_mask=0, data_size=1 (2 bytes)
             (int)0x01000000,   // tensor_dim0[15:0]=256 in bits[63:48]
             (int)0x01000000,   // tensor_dim1[15:0]=256 in bits[95:80]
             (int)0x01000000,   // tile_dim0=256 in bits[127:112]
             16,                // tile_dim1=16 rows
             256,               // tensor_dim0_stride (elements)
             0, 0 };
  v4i gz = { 0, 0, 0, 0 };
#if __has_include(<hip/amd_detail/amd_gfx1250_TDM.h>)
  v8i gz8 = { 0, 0, 0, 0, 0, 0, 0, 0 };
  __builtin_amdgcn_tensor_load_to_lds(g0, g1, gz, gz, gz8, 0);
#else
  __builtin_amdgcn_tensor_load_to_lds(g0, g1, gz, gz, 0);
#endif
}

// ---------------- WMMA accumulate: A from global, B tile from LDS ----------
__device__ __forceinline__ void wmma_accum_pair_lds(
    const unsigned short* __restrict__ Abf,
    const unsigned short* Bsm,           // LDS tile: 16 rows x 256 K, bf16
    int m0, int r, int hi,
    CFrag& acc0, CFrag& acc1) {
  #pragma unroll
  for (int k0 = 0; k0 < EMB_C; k0 += 32) {
    AFrag a0, a1, b;
    #pragma unroll
    for (int p = 0; p < 8; ++p) {
      // ISA 7.12.2 16-bit A/B fragment K mapping per lane half
      int kp = ((p < 4) ? (2 * p) : (2 * p + 8)) + hi * 8;
      b.u[p]  = *(const unsigned int*)(Bsm + (size_t)r * EMB_C + k0 + kp);
      a0.u[p] = *(const unsigned int*)(Abf + (size_t)(m0 + r) * EMB_C + k0 + kp);
      a1.u[p] = *(const unsigned int*)(Abf + (size_t)(m0 + 16 + r) * EMB_C + k0 + kp);
    }
    acc0.v = __builtin_amdgcn_wmma_f32_16x16x32_bf16(
        false, a0.v, false, b.v, (short)0, acc0.v, false, false);
    acc1.v = __builtin_amdgcn_wmma_f32_16x16x32_bf16(
        false, a1.v, false, b.v, (short)0, acc1.v, false, false);
  }
}

// out = meanB @ Wl^T + b + xB @ Wr^T  (+ optional ReLU)
// block = one 16-wide N tile; wave 0 TDM-stages the two 16x256 W tiles to LDS;
// each of 8 waves then computes two 16x16 M tiles against the shared B tile.
__global__ void sage_gemm(const unsigned short* __restrict__ meanB,
                          const unsigned short* __restrict__ xB,
                          const unsigned short* __restrict__ Wl,
                          const unsigned short* __restrict__ Wr,
                          const float* __restrict__ bias,
                          float* __restrict__ out,
                          int do_relu) {
  __shared__ unsigned short sW[2 * 16 * EMB_C];   // 16 KB: Wl tile | Wr tile

  int n_tile = blockIdx.x & 15;
  int mg     = blockIdx.x >> 4;          // 0..390
  int wid    = threadIdx.x >> 5;
  int lane   = threadIdx.x & 31;
  int r  = lane & 15;
  int hi = lane >> 4;

  if (wid == 0) {
    unsigned base = (unsigned)(unsigned long long)sW;
    tdm_load_16x256_bf16(Wl + (size_t)n_tile * 16 * EMB_C, base);
    tdm_load_16x256_bf16(Wr + (size_t)n_tile * 16 * EMB_C, base + 16u * EMB_C * 2u);
    __builtin_amdgcn_s_wait_tensorcnt(0);
  }
  __syncthreads();

  int mpair = mg * 8 + wid;              // 0..3124 valid
  if (mpair < 3125) {
    int m0   = mpair * 32;
    int ncol = n_tile * 16 + r;

    CFrag acc0, acc1;
    #pragma unroll
    for (int v = 0; v < 8; ++v) { acc0.f[v] = 0.0f; acc1.f[v] = 0.0f; }

    wmma_accum_pair_lds(meanB, sW,                m0, r, hi, acc0, acc1);
    wmma_accum_pair_lds(xB,    sW + 16 * EMB_C,   m0, r, hi, acc0, acc1);

    float bv = bias[ncol];
    #pragma unroll
    for (int v = 0; v < 8; ++v) {
      int row = m0 + v + hi * 8;                  // C/D layout
      float v0 = acc0.f[v] + bv;
      float v1 = acc1.f[v] + bv;
      if (do_relu) { v0 = fmaxf(v0, 0.0f); v1 = fmaxf(v1, 0.0f); }
      out[(size_t)row * EMB_C + ncol]        = v0;
      out[(size_t)(row + 16) * EMB_C + ncol] = v1;
    }
  }
}

// ---------------- supervision edge dot products ----------------
__global__ void dot_scores(const float* __restrict__ x,
                           const int* __restrict__ s,
                           const int* __restrict__ d,
                           float* __restrict__ out) {
  int wave = (blockIdx.x * blockDim.x + threadIdx.x) >> 5;
  int lane = threadIdx.x & 31;
  if (wave >= N_SUP_E) return;
  size_t si = (size_t)s[wave] * EMB_C;
  size_t di = (size_t)d[wave] * EMB_C;
  float acc = 0.0f;
  #pragma unroll
  for (int c = 0; c < EMB_C; c += 32)
    acc += x[si + c + lane] * x[di + c + lane];
  #pragma unroll
  for (int off = 16; off > 0; off >>= 1)
    acc += __shfl_xor(acc, off, 32);
  if (lane == 0) out[wave] = acc;
}

extern "C" void kernel_launch(void* const* d_in, const int* in_sizes, int n_in,
                              void* d_out, int out_size, void* d_ws, size_t ws_size,
                              hipStream_t stream) {
  (void)in_sizes; (void)n_in; (void)out_size; (void)ws_size;

  const float* emb = (const float*)d_in[0];
  const int*   mp  = (const int*)d_in[1];
  const int*   sup = (const int*)d_in[2];
  const float* Wl[3] = {(const float*)d_in[3], (const float*)d_in[6], (const float*)d_in[9]};
  const float* bl[3] = {(const float*)d_in[4], (const float*)d_in[7], (const float*)d_in[10]};
  const float* Wr[3] = {(const float*)d_in[5], (const float*)d_in[8], (const float*)d_in[11]};

  const int* src  = mp;
  const int* dstp = mp + N_MP_E;
  const int* ss   = sup;
  const int* sd   = sup + N_SUP_E;

  char* p = (char*)d_ws;
  auto take = [&](size_t bytes) -> char* {
    char* q = p; p += (bytes + 255) & ~(size_t)255; return q;
  };
  const size_t NF = (size_t)N_NODES_C * EMB_C;
  float*          sum   = (float*)take(NF * sizeof(float));
  float*          cnt   = (float*)take((size_t)N_NODES_C * sizeof(float));
  float*          xping = (float*)take(NF * sizeof(float));
  float*          xpong = (float*)take(NF * sizeof(float));
  unsigned short* meanB = (unsigned short*)take(NF * 2);
  unsigned short* xB    = (unsigned short*)take(NF * 2);
  unsigned short* wbf   = (unsigned short*)take(6ull * EMB_C * EMB_C * 2);

  // convert all weight matrices to bf16 once
  for (int l = 0; l < 3; ++l) {
    cvt_f32_bf16<<<(EMB_C * EMB_C) / 256, 256, 0, stream>>>(
        Wl[l], wbf + (size_t)(2 * l + 0) * EMB_C * EMB_C, EMB_C * EMB_C);
    cvt_f32_bf16<<<(EMB_C * EMB_C) / 256, 256, 0, stream>>>(
        Wr[l], wbf + (size_t)(2 * l + 1) * EMB_C * EMB_C, EMB_C * EMB_C);
  }

  const float* xin  = emb;
  float*       xout = xping;
  for (int l = 0; l < 3; ++l) {
    hipMemsetAsync(sum, 0, NF * sizeof(float), stream);
    hipMemsetAsync(cnt, 0, (size_t)N_NODES_C * sizeof(float), stream);
    sage_scatter<<<(N_MP_E + 7) / 8, 256, 0, stream>>>(xin, src, dstp, sum, cnt);
    mean_cvt<<<(int)(NF / 256), 256, 0, stream>>>(sum, cnt, xin, meanB, xB);
    // 16 n-tiles * 391 m-groups (8 waves/block, mpair guard at 3125)
    sage_gemm<<<16 * 391, 256, 0, stream>>>(
        meanB, xB,
        wbf + (size_t)(2 * l + 0) * EMB_C * EMB_C,
        wbf + (size_t)(2 * l + 1) * EMB_C * EMB_C,
        bl[l], xout, (l < 2) ? 1 : 0);
    xin  = xout;
    xout = (xout == xping) ? xpong : xping;
  }

  dot_scores<<<N_SUP_E / 8, 256, 0, stream>>>(xin, ss, sd, (float*)d_out);
}
